// VQVAE_55903294324898
// MI455X (gfx1250) — matrix operand order repaired and checked by reading, not compiled, not run on hardware
//
#include <hip/hip_runtime.h>

// ---------------------------------------------------------------------------
// VQ-VAE forward for MI455X (gfx1250).
// BF16 WMMA implicit-GEMM; double-buffered TDM (tensor_load_to_lds) pipeline
// for plain tiles; branch-free vectorized gathers for conv/convT tiles.
// ---------------------------------------------------------------------------

#define BM 64
#define BN 64
#define BK 32
#define LDST 40   // LDS row stride in ushort (80B: 16B-aligned, conflict-free)
#define TSZ (BM * LDST)

typedef __attribute__((ext_vector_type(16))) __bf16          bf16x16;
typedef __attribute__((ext_vector_type(8)))  float           floatx8;
typedef __attribute__((ext_vector_type(8)))  unsigned short  ushort8;

union FragAB { unsigned short u[16]; ushort8 h[2]; bf16x16 v; };
union FragC  { float f[8]; floatx8 v; };
union Pack16 { unsigned short u[16]; ushort8 h[2]; };

__device__ __forceinline__ unsigned short f2bf(float f) {
  unsigned int x = __float_as_uint(f);
  unsigned int r = x + 0x7fffu + ((x >> 16) & 1u);   // round-to-nearest-even
  return (unsigned short)(r >> 16);
}

#if __has_builtin(__builtin_amdgcn_tensor_load_to_lds)
#define HAVE_TDM 1
typedef __attribute__((ext_vector_type(4))) unsigned int uint4v;
typedef __attribute__((ext_vector_type(8))) int          int8v;
typedef __attribute__((ext_vector_type(4))) int          int4v;

// TDM: async-copy a 16-row x 32-elem bf16 tile (row stride K elems) into LDS,
// padding 16B after every 64B row (-> 80B LDS rows). Completion via TENSORcnt.
__device__ __forceinline__ void tdm_tile16(unsigned lds_byte_off,
                                           const unsigned short* gptr, int K) {
  unsigned long long ga = (unsigned long long)gptr;
  uint4v g0;
  g0[0] = 1u;                                    // count=1
  g0[1] = lds_byte_off;                          // lds_addr
  g0[2] = (unsigned)ga;                          // global_addr[31:0]
  g0[3] = (unsigned)((ga >> 32) & 0x01ffffffu) | (2u << 30);  // addr hi | type=2
  int8v g1;
  g1[0] = (int)((1u << 16) | (1u << 20) | (3u << 22) | (3u << 25));
  //        data_size=2B    pad_enable   interval=16dw  amount=4dw
  g1[1] = (int)(32u << 16);                      // tensor_dim0 = 32
  g1[2] = (int)(16u << 16);                      // tensor_dim1 = 16
  g1[3] = (int)(32u << 16);                      // tile_dim0 = 32
  g1[4] = 16;                                    // tile_dim1 = 16, tile_dim2 = 0
  g1[5] = K;                                     // tensor_dim0_stride[31:0]
  g1[6] = 0;
  g1[7] = 0;
  int4v gz; gz[0] = gz[1] = gz[2] = gz[3] = 0;
#if __clang_major__ >= 23
  int8v gz8; gz8[0]=gz8[1]=gz8[2]=gz8[3]=gz8[4]=gz8[5]=gz8[6]=gz8[7]=0;
  __builtin_amdgcn_tensor_load_to_lds(g0, g1, gz, gz, gz8, 0);
#else
  __builtin_amdgcn_tensor_load_to_lds(g0, g1, gz, gz, 0);
#endif
}
#endif

// ---------------------------------------------------------------------------
// Templated WMMA GEMM: C[M,N] = A[M,K] * B[K,N] (+bias, opt ReLU). All bf16 in.
// AM: 0 plain row-major A (TDM pipeline); 1 conv fwd im2col (s2 p1 k4, K=Cin*16)
//     2 convT gather for parity (py,px), K=Cin*4, rows over half-dims
// BMD:0 weights [N][K] row-major (TDM pipeline); 2 convT weights (Cin,Cout,4,4)
// OM: 0 row-major; 1 NCHW scatter; 2 NCHW scatter with parity
// OF32: write f32 (else bf16).   Requires M%64==0, K%32==0.
// Double-buffered LDS: tile k+1 staged (TDM or gather) under tile-k WMMAs.
// ---------------------------------------------------------------------------
template<int AM, int BMD, int OM, bool OF32>
__global__ __launch_bounds__(128) void gemm_wmma(
    const unsigned short* __restrict__ Asrc,
    const unsigned short* __restrict__ Bsrc,
    const float* __restrict__ bias, void* __restrict__ Cout,
    int M, int N, int K, int relu,
    int Cin, int IH, int IW, int lg_ow, int lg_sp, int CoutCh, int py, int px)
{
  __shared__ unsigned short As[2 * TSZ];
  __shared__ unsigned short Bs[2 * TSZ];

  const int tid  = threadIdx.x;
  const int lane = tid & 31;
  const int wave = tid >> 5;
  const int m0   = blockIdx.y * BM;
  const int n0   = blockIdx.x * BN;
  const int wy   = (wave >> 1) * 32;
  const int wx   = (wave & 1) * 32;
  const int half = lane >> 4;
  const int l15  = lane & 15;
  const int OW   = 1 << lg_ow;
  const int spm  = (1 << lg_sp) - 1;

#ifdef HAVE_TDM
  const unsigned asoff = (unsigned)(unsigned long long)&As[0];
  const unsigned bsoff = (unsigned)(unsigned long long)&Bs[0];
  const int wrow = __builtin_amdgcn_readfirstlane(wave * 16);
#endif

  // ---- per-thread staging constants (one 16-elem half-row per thread) ----
  const int sr  = tid >> 1;            // tile row / tile col (0..63)
  const int kk0 = (tid & 1) * 16;      // k-half within the 32-wide tile
  const int a_gr  = m0 + sr;
  const int a_n   = a_gr >> lg_sp;
  const int a_rem = a_gr & spm;
  const int a_oy  = a_rem >> lg_ow;
  const int a_ox  = a_rem & (OW - 1);
  const int b_gc  = n0 + sr;

  // ---- branch-free gather staging: clamp addr, load, select, 2x b128 store --
  auto stageA = [&](unsigned short* dst, int k0) {
    Pack16 pk;
    if (AM == 1) {
      const int ci = (k0 + kk0) >> 4;               // constant per half
      const unsigned short* base = Asrc + ((long)a_n * Cin + ci) * IH * IW;
#pragma unroll
      for (int ky = 0; ky < 4; ++ky) {
        const int iy  = a_oy * 2 - 1 + ky;
        const bool oky = (iy >= 0) && (iy < IH);
        const int cy  = oky ? iy : 0;
#pragma unroll
        for (int kx = 0; kx < 4; ++kx) {
          const int ix  = a_ox * 2 - 1 + kx;
          const bool okx = (ix >= 0) && (ix < IW);
          const int cx  = okx ? ix : 0;
          const unsigned short t = base[cy * IW + cx];   // always in-bounds
          pk.u[ky * 4 + kx] = (oky && okx) ? t : (unsigned short)0;
        }
      }
    } else {  // AM == 2
      const int ci0 = (k0 + kk0) >> 2;              // 4 channels per half
#pragma unroll
      for (int c4 = 0; c4 < 4; ++c4) {
        const unsigned short* base = Asrc + ((long)a_n * Cin + ci0 + c4) * IH * IW;
#pragma unroll
        for (int ty = 0; ty < 2; ++ty) {
          const int iy  = a_oy + py - ty;
          const bool oky = (iy >= 0) && (iy < IH);
          const int cy  = oky ? iy : 0;
#pragma unroll
          for (int tx = 0; tx < 2; ++tx) {
            const int ix  = a_ox + px - tx;
            const bool okx = (ix >= 0) && (ix < IW);
            const int cx  = okx ? ix : 0;
            const unsigned short t = base[cy * IW + cx];
            pk.u[c4 * 4 + ty * 2 + tx] = (oky && okx) ? t : (unsigned short)0;
          }
        }
      }
    }
    unsigned short* d = dst + sr * LDST + kk0;      // 16B-aligned
    *(ushort8*)(d)     = pk.h[0];
    *(ushort8*)(d + 8) = pk.h[1];
  };
  auto stageB = [&](unsigned short* dst, int k0) {   // BMD == 2 only
    Pack16 pk;
    const int ci0 = (k0 + kk0) >> 2;
    const bool okc = b_gc < N;
    const int cc  = okc ? b_gc : (N - 1);            // clamp for safe loads
#pragma unroll
    for (int c4 = 0; c4 < 4; ++c4) {
      const unsigned short* base = Bsrc + ((long)(ci0 + c4) * CoutCh + cc) * 16;
#pragma unroll
      for (int ty = 0; ty < 2; ++ty) {
        const int ky = (1 - py) + 2 * ty;
#pragma unroll
        for (int tx = 0; tx < 2; ++tx) {
          const int kx = (1 - px) + 2 * tx;
          const unsigned short t = base[ky * 4 + kx];
          pk.u[c4 * 4 + ty * 2 + tx] = okc ? t : (unsigned short)0;
        }
      }
    }
    unsigned short* d = dst + sr * LDST + kk0;
    *(ushort8*)(d)     = pk.h[0];
    *(ushort8*)(d + 8) = pk.h[1];
  };
#ifndef HAVE_TDM
  auto copyA = [&](unsigned short* dst, int k0) {
    for (int i = tid; i < (BM * BK) / 8; i += 128) {
      const int r = i >> 2, kc = (i & 3) * 8;
      *(ushort8*)&dst[r * LDST + kc] =
          *(const ushort8*)&Asrc[(long)(m0 + r) * K + k0 + kc];
    }
  };
  auto copyB = [&](unsigned short* dst, int k0) {
    for (int i = tid; i < (BN * BK) / 8; i += 128) {
      const int c = i >> 2, kc = (i & 3) * 8;
      *(ushort8*)&dst[c * LDST + kc] =
          *(const ushort8*)&Bsrc[(long)(n0 + c) * K + k0 + kc];
    }
  };
#endif

  FragC acc[2][2];
#pragma unroll
  for (int mi = 0; mi < 2; ++mi)
#pragma unroll
    for (int ni = 0; ni < 2; ++ni)
#pragma unroll
      for (int j = 0; j < 8; ++j) acc[mi][ni].f[j] = 0.0f;

  // ---- prologue: stage tile 0 into buffer 0 ----
#ifdef HAVE_TDM
  if (AM == 0)  tdm_tile16(asoff + (unsigned)(wrow * LDST * 2),
                           Asrc + (long)(m0 + wrow) * K, K);
  if (BMD == 0) tdm_tile16(bsoff + (unsigned)(wrow * LDST * 2),
                           Bsrc + (long)(n0 + wrow) * K, K);
#else
  if (AM == 0)  copyA(As, 0);
  if (BMD == 0) copyB(Bs, 0);
#endif
  if (AM != 0)  stageA(As, 0);
  if (BMD != 0) stageB(Bs, 0);
#ifdef HAVE_TDM
  if (AM == 0 || BMD == 0) __builtin_amdgcn_s_wait_tensorcnt(0);
#endif
  __syncthreads();

  const int aoff0 = half ? 8 : 0;
  const int aoff1 = half ? 24 : 16;
  const int boff  = half ? 8 : 0;

  int cur = 0;
  for (int k0 = 0; k0 < K; k0 += BK) {
    const int nxt = cur ^ 1;
    const bool more = (k0 + BK) < K;
    unsigned short* An = As + nxt * TSZ;
    unsigned short* Bn = Bs + nxt * TSZ;

    // issue next-tile TDMs first so the DMA runs under this tile's math
#ifdef HAVE_TDM
    if (more) {
      if (AM == 0)
        tdm_tile16(asoff + (unsigned)(nxt * TSZ * 2 + wrow * LDST * 2),
                   Asrc + (long)(m0 + wrow) * K + (k0 + BK), K);
      if (BMD == 0)
        tdm_tile16(bsoff + (unsigned)(nxt * TSZ * 2 + wrow * LDST * 2),
                   Bsrc + (long)(n0 + wrow) * K + (k0 + BK), K);
    }
#endif

    // fragment loads from current buffer (ds_load_b128, ISA bf16 striping)
    const unsigned short* Ac = As + cur * TSZ;
    const unsigned short* Bc = Bs + cur * TSZ;
    FragAB afr[2], bfr[2];
#pragma unroll
    for (int mi = 0; mi < 2; ++mi) {
      const unsigned short* p = &Ac[(wy + mi * 16 + l15) * LDST];
      afr[mi].h[0] = *(const ushort8*)(p + aoff0);
      afr[mi].h[1] = *(const ushort8*)(p + aoff1);
    }
#pragma unroll
    for (int ni = 0; ni < 2; ++ni) {
      const unsigned short* p = &Bc[(wx + ni * 16 + l15) * LDST];
      bfr[ni].h[0] = *(const ushort8*)(p + boff);
      bfr[ni].h[1] = *(const ushort8*)(p + boff + 16);
    }

    // manual staging of next tile (writes the other buffer)
    if (more) {
#ifndef HAVE_TDM
      if (AM == 0)  copyA(An, k0 + BK);
      if (BMD == 0) copyB(Bn, k0 + BK);
#endif
      if (AM != 0)  stageA(An, k0 + BK);
      if (BMD != 0) stageB(Bn, k0 + BK);
    }

#pragma unroll
    for (int mi = 0; mi < 2; ++mi)
#pragma unroll
      for (int ni = 0; ni < 2; ++ni)
        acc[mi][ni].v = __builtin_amdgcn_wmma_f32_16x16x32_bf16(
            false, afr[mi].v, false, bfr[ni].v, (short)0, acc[mi][ni].v,
            false, false);

#ifdef HAVE_TDM
    if ((AM == 0 || BMD == 0) && more) __builtin_amdgcn_s_wait_tensorcnt(0);
#endif
    __syncthreads();
    cur = nxt;
  }

  // ---- epilogue: bias, ReLU, scatter ----
#pragma unroll
  for (int mi = 0; mi < 2; ++mi) {
#pragma unroll
    for (int ni = 0; ni < 2; ++ni) {
      const int col = n0 + wx + ni * 16 + l15;
      if (col >= N) continue;
      const float bv = bias[col];
#pragma unroll
      for (int j = 0; j < 8; ++j) {
        const int row = m0 + wy + mi * 16 + half * 8 + j;
        float v = acc[mi][ni].f[j] + bv;
        if (relu) v = fmaxf(v, 0.0f);
        long off;
        if (OM == 0) {
          off = (long)row * N + col;
        } else {
          const int n = row >> lg_sp, rem = row & spm;
          if (OM == 1) {
            off = (((long)n * CoutCh + col) << lg_sp) + rem;
          } else {
            const int oy2 = rem >> lg_ow, ox2 = rem & (OW - 1);
            off = (((long)n * CoutCh + col) << (lg_sp + 2)) +
                  ((long)(oy2 * 2 + py) << (lg_ow + 1)) + (ox2 * 2 + px);
          }
        }
        if (OF32) ((float*)Cout)[off] = v;
        else      ((unsigned short*)Cout)[off] = f2bf(v);
      }
    }
  }
}

// ---------------------------------------------------------------------------
// conv0: x(256,3,64,64) f32 -> h0(256,64,32,32) bf16, stride2 pad1 k4, ReLU.
// ---------------------------------------------------------------------------
__global__ __launch_bounds__(256) void conv0_kernel(
    const float* __restrict__ x, const float* __restrict__ w,
    const float* __restrict__ b, unsigned short* __restrict__ out)
{
  __shared__ float ws[64 * 48];
  for (int i = threadIdx.x; i < 64 * 48; i += 256) ws[i] = w[i];
  __syncthreads();
  const int idx = blockIdx.x * 256 + threadIdx.x;
  const int ox = idx & 31, oy = (idx >> 5) & 31;
  const int co = (idx >> 10) & 63, n = idx >> 16;
  float acc = b[co];
#pragma unroll
  for (int ci = 0; ci < 3; ++ci) {
#pragma unroll
    for (int ky = 0; ky < 4; ++ky) {
      const int iy = oy * 2 - 1 + ky;
      const bool oky = (iy >= 0) && (iy < 64);
      const int cy = oky ? iy : 0;
#pragma unroll
      for (int kx = 0; kx < 4; ++kx) {
        const int ix = ox * 2 - 1 + kx;
        const bool okx = (ix >= 0) && (ix < 64);
        const int cx = okx ? ix : 0;
        const float t = x[(((long)n * 3 + ci) * 64 + cy) * 64 + cx];
        acc = fmaf((oky && okx) ? t : 0.0f,
                   ws[co * 48 + ci * 16 + ky * 4 + kx], acc);
      }
    }
  }
  out[idx] = f2bf(fmaxf(acc, 0.0f));
}

// f32 -> bf16, keep layout
__global__ void cvt_kernel(const float* __restrict__ in,
                           unsigned short* __restrict__ out, int n) {
  const int i = blockIdx.x * 256 + threadIdx.x;
  if (i < n) out[i] = f2bf(in[i]);
}

// f32 [K][N] -> bf16 [N][K]
__global__ void cvt_t_kernel(const float* __restrict__ in,
                             unsigned short* __restrict__ out, int K, int N) {
  const int i = blockIdx.x * 256 + threadIdx.x;
  if (i < K * N) {
    const int n = i / K, k = i - n * K;
    out[i] = f2bf(in[(long)k * N + n]);
  }
}

// ---------------------------------------------------------------------------
// VQ: one wave32 per 64-dim latent chunk; shfl_xor reductions; argmin-first;
// straight-through into bf16 e; 1.25*mean(diff^2) atomically into loss.
// ---------------------------------------------------------------------------
__global__ __launch_bounds__(256) void vq_kernel(
    const float* __restrict__ z, const float* __restrict__ cb,
    unsigned short* __restrict__ e, float* __restrict__ loss)
{
  const int wave = (blockIdx.x * 256 + threadIdx.x) >> 5;   // 0..4095
  const int lane = threadIdx.x & 31;
  const int n = wave >> 4, slot = wave & 15;
  const float* zp = z + (long)n * 1536 + slot * 64;
  const float z0 = zp[lane * 2 + 0];
  const float z1 = zp[lane * 2 + 1];
  float best = 3.0e38f;
  int bidx = 0;
  for (int c = 0; c < 512; ++c) {
    const float* cp = cb + c * 64;
    const float c0 = cp[lane * 2 + 0];
    const float c1 = cp[lane * 2 + 1];
    float d = (z0 - c0) * (z0 - c0) + (z1 - c1) * (z1 - c1);
#pragma unroll
    for (int off = 16; off > 0; off >>= 1) d += __shfl_xor(d, off, 32);
    if (d < best) { best = d; bidx = c; }
  }
  const float* cp = cb + bidx * 64;
  const float q0 = cp[lane * 2 + 0];
  const float q1 = cp[lane * 2 + 1];
  e[(long)n * 1536 + slot * 64 + lane * 2 + 0] = f2bf(q0);
  e[(long)n * 1536 + slot * 64 + lane * 2 + 1] = f2bf(q1);
  float s = (q0 - z0) * (q0 - z0) + (q1 - z1) * (q1 - z1);
#pragma unroll
  for (int off = 16; off > 0; off >>= 1) s += __shfl_xor(s, off, 32);
  if (lane == 0) atomicAdd(loss, s * (1.25f / 262144.0f));
}

__global__ void zero_loss_kernel(float* __restrict__ p) {
  if (threadIdx.x == 0 && blockIdx.x == 0) *p = 0.0f;
}

// ---------------------------------------------------------------------------
extern "C" void kernel_launch(void* const* d_in, const int* in_sizes, int n_in,
                              void* d_out, int out_size, void* d_ws, size_t ws_size,
                              hipStream_t stream) {
  (void)in_sizes; (void)n_in; (void)out_size; (void)ws_size;

  const float* x    = (const float*)d_in[0];
  const float* ew0  = (const float*)d_in[1];  const float* eb0 = (const float*)d_in[2];
  const float* ew1  = (const float*)d_in[3];  const float* eb1 = (const float*)d_in[4];
  const float* ew2  = (const float*)d_in[5];  const float* eb2 = (const float*)d_in[6];
  const float* ew3  = (const float*)d_in[7];  const float* eb3 = (const float*)d_in[8];
  const float* fw0  = (const float*)d_in[9];  const float* fb0 = (const float*)d_in[10];
  const float* fw1  = (const float*)d_in[11]; const float* fb1 = (const float*)d_in[12];
  const float* fw2  = (const float*)d_in[13]; const float* fb2 = (const float*)d_in[14];
  const float* cb   = (const float*)d_in[15];
  const float* gw0  = (const float*)d_in[16]; const float* gb0 = (const float*)d_in[17];
  const float* gw1  = (const float*)d_in[18]; const float* gb1 = (const float*)d_in[19];
  const float* gw2  = (const float*)d_in[20]; const float* gb2 = (const float*)d_in[21];
  const float* tw0  = (const float*)d_in[22]; const float* tb0 = (const float*)d_in[23];
  const float* tw1  = (const float*)d_in[24]; const float* tb1 = (const float*)d_in[25];
  const float* tw2  = (const float*)d_in[26]; const float* tb2 = (const float*)d_in[27];
  const float* tw3  = (const float*)d_in[28]; const float* tb3 = (const float*)d_in[29];

  float* recon = (float*)d_out;                 // 256*3*64*64 f32
  float* loss  = recon + 3145728;

  // ---- workspace carve (bf16 ushorts, then f32 z) ----
  unsigned short* p = (unsigned short*)d_ws;
  auto take = [&](long n) { unsigned short* r = p; p += n; return r; };
  unsigned short* ew1b = take(64 * 1024);
  unsigned short* ew2b = take(128 * 1024);
  unsigned short* ew3b = take(256 * 2048);
  unsigned short* fw0t = take((long)1024 * 4096);
  unsigned short* fw1t = take((long)1024 * 1024);
  unsigned short* fw2t = take((long)1536 * 1024);
  unsigned short* gw0t = take((long)512 * 1536);
  unsigned short* gw1t = take((long)1024 * 512);
  unsigned short* gw2t = take((long)4096 * 1024);
  unsigned short* tw0b = take(256 * 128 * 16);
  unsigned short* tw1b = take(128 * 64 * 16);
  unsigned short* tw2b = take(64 * 64 * 16);
  unsigned short* tw3b = take(64 * 3 * 16 + 16);
  unsigned short* h0 = take(16777216);
  unsigned short* h1 = take(4194304);
  unsigned short* h2 = take(2097152);
  unsigned short* h3 = take(1048576);
  unsigned short* f0 = take(262144);
  unsigned short* f1 = take(262144);
  unsigned short* e  = take(393216);
  unsigned short* g0 = take(131072);
  unsigned short* g1 = take(262144);
  unsigned short* g2 = take(1048576);
  unsigned short* d0 = take(2097152);
  unsigned short* d1 = take(4194304);
  unsigned short* d2 = take(16777216);
  float* z = (float*)(p + ((unsigned long long)p & 1));   // f32, 256*1536

  auto cvt = [&](const float* in, unsigned short* out, long n) {
    cvt_kernel<<<(unsigned)((n + 255) / 256), 256, 0, stream>>>(in, out, (int)n);
  };
  auto cvtT = [&](const float* in, unsigned short* out, int K, int N) {
    cvt_t_kernel<<<(unsigned)(((long)K * N + 255) / 256), 256, 0, stream>>>(in, out, K, N);
  };

  // ---- weight prep (f32 -> bf16; FC weights transposed to [N][K]) ----
  cvt(ew1, ew1b, 64 * 1024);
  cvt(ew2, ew2b, 128 * 1024);
  cvt(ew3, ew3b, 256 * 2048);
  cvtT(fw0, fw0t, 4096, 1024);
  cvtT(fw1, fw1t, 1024, 1024);
  cvtT(fw2, fw2t, 1024, 1536);
  cvtT(gw0, gw0t, 1536, 512);
  cvtT(gw1, gw1t, 512, 1024);
  cvtT(gw2, gw2t, 1024, 4096);
  cvt(tw0, tw0b, 256 * 128 * 16);
  cvt(tw1, tw1b, 128 * 64 * 16);
  cvt(tw2, tw2b, 64 * 64 * 16);
  cvt(tw3, tw3b, 64 * 3 * 16);

#define GEMM(AM, BMD, OM, OF32, A, B, BIAS, C, M, N, K, RELU, CIN, IH, IW, LGOW, LGSP, CO, PY, PX)      \
  gemm_wmma<AM, BMD, OM, OF32><<<dim3(((N) + 63) / 64, (M) / 64), 128, 0, stream>>>(                     \
      A, B, BIAS, C, M, N, K, RELU, CIN, IH, IW, LGOW, LGSP, CO, PY, PX)

  // ---- encoder ----
  conv0_kernel<<<65536, 256, 0, stream>>>(x, ew0, eb0, h0);
  GEMM(1, 0, 1, false, h0, ew1b, eb1, h1, 65536,  64, 1024, 1,  64, 32, 32, 4, 8,  64, 0, 0);
  GEMM(1, 0, 1, false, h1, ew2b, eb2, h2, 16384, 128, 1024, 1,  64, 16, 16, 3, 6, 128, 0, 0);
  GEMM(1, 0, 1, false, h2, ew3b, eb3, h3,  4096, 256, 2048, 0, 128,  8,  8, 2, 4, 256, 0, 0);
  GEMM(0, 0, 0, false, h3, fw0t, fb0, f0,   256, 1024, 4096, 1, 0, 0, 0, 0, 0, 0, 0, 0);
  GEMM(0, 0, 0, false, f0, fw1t, fb1, f1,   256, 1024, 1024, 1, 0, 0, 0, 0, 0, 0, 0, 0);
  GEMM(0, 0, 0, true,  f1, fw2t, fb2, z,    256, 1536, 1024, 0, 0, 0, 0, 0, 0, 0, 0, 0);

  // ---- vector quantize ----
  cvt(z, e, 393216);                       // seed e with bf16(z); VQ overwrites z_c part
  zero_loss_kernel<<<1, 32, 0, stream>>>(loss);
  vq_kernel<<<512, 256, 0, stream>>>(z, cb, e, loss);

  // ---- decoder FCs ----
  GEMM(0, 0, 0, false, e,  gw0t, gb0, g0, 256,  512, 1536, 1, 0, 0, 0, 0, 0, 0, 0, 0);
  GEMM(0, 0, 0, false, g0, gw1t, gb1, g1, 256, 1024,  512, 1, 0, 0, 0, 0, 0, 0, 0, 0);
  GEMM(0, 0, 0, false, g1, gw2t, gb2, g2, 256, 4096, 1024, 0, 0, 0, 0, 0, 0, 0, 0, 0);

  // ---- transposed convs: 4 exact GEMMs per layer (one per output parity) ----
  for (int py = 0; py < 2; ++py)
    for (int px = 0; px < 2; ++px) {
      GEMM(2, 2, 2, false, g2, tw0b, tb0, d0,    4096, 128, 1024, 1, 256,  4,  4, 2,  4, 128, py, px);
      GEMM(2, 2, 2, false, d0, tw1b, tb1, d1,   16384,  64,  512, 1, 128,  8,  8, 3,  6,  64, py, px);
      GEMM(2, 2, 2, false, d1, tw2b, tb2, d2,   65536,  64,  256, 1,  64, 16, 16, 4,  8,  64, py, px);
      GEMM(2, 2, 2, true,  d2, tw3b, tb3, recon, 262144, 3,  256, 0,  64, 32, 32, 5, 10,   3, py, px);
    }
#undef GEMM
}
